// EnergyGNN_71683004170820
// MI455X (gfx1250) — matrix operand order repaired and checked by reading, not compiled, not run on hardware
//
#include <hip/hip_runtime.h>
#include <hip/hip_bf16.h>
#include <math.h>

#define NNODES   50000
#define NEDGES   1600000
#define EPRIME   (NEDGES + NNODES)     // edges + self loops
#define DIM      128
#define NH       4
#define DPH      32
#define NEG_SLOPE 0.2f
#define LN_EPS    1e-5f
#define LDH      136                   // padded LDS half-stride (keeps 16B alignment)

typedef __attribute__((ext_vector_type(16))) _Float16 v16h;
typedef __attribute__((ext_vector_type(8)))  _Float16 v8h;
typedef __attribute__((ext_vector_type(8)))  float    v8f;

__device__ __forceinline__ void edge_endpoints(const int* __restrict__ ei, int e,
                                               int& s, int& d) {
    if (e < NEDGES) { s = ei[e]; d = ei[NEDGES + e]; }
    else            { s = e - NEDGES; d = s; }       // self loop
}

// ---------------------------------------------------------------- h0 = x @ w_in + b_in
__global__ __launch_bounds__(256)
void k_input(const float* __restrict__ x, const float* __restrict__ w_in,
             const float* __restrict__ b_in, float* __restrict__ h) {
    int i = blockIdx.x * 256 + threadIdx.x;
    if (i >= NNODES * DIM) return;
    int n = i >> 7, j = i & 127;
    const float* xr = x + n * 5;
    float s = b_in[j];
    #pragma unroll
    for (int k = 0; k < 5; ++k) s += xr[k] * w_in[k * DIM + j];
    h[i] = s;
}

// ---------------------------------------------------------------- CSR build (once per call)
__global__ __launch_bounds__(256)
void k_csr_init(int* __restrict__ deg, int* __restrict__ cursor) {
    int i = blockIdx.x * 256 + threadIdx.x;
    if (i < NNODES) { deg[i] = 0; cursor[i] = 0; }
}

__global__ __launch_bounds__(256)
void k_csr_count(const int* __restrict__ ei, int* __restrict__ deg) {
    int e = blockIdx.x * 256 + threadIdx.x;
    if (e >= EPRIME) return;
    int s, d; edge_endpoints(ei, e, s, d); (void)s;
    atomicAdd(&deg[d], 1);
}

// single-block exclusive scan (Hillis-Steele over 1024-chunks, serial carry)
__global__ __launch_bounds__(1024)
void k_csr_scan(const int* __restrict__ deg, int* __restrict__ rowstart) {
    __shared__ int sdata[1024];
    __shared__ int s_carry;
    const int t = threadIdx.x;
    if (t == 0) s_carry = 0;
    __syncthreads();
    for (int base = 0; base < NNODES; base += 1024) {
        int v = (base + t < NNODES) ? deg[base + t] : 0;
        sdata[t] = v;
        __syncthreads();
        #pragma unroll
        for (int off = 1; off < 1024; off <<= 1) {
            int add = (t >= off) ? sdata[t - off] : 0;
            __syncthreads();
            sdata[t] += add;
            __syncthreads();
        }
        if (base + t < NNODES) rowstart[base + t] = s_carry + (sdata[t] - v);
        __syncthreads();
        if (t == 0) s_carry += sdata[1023];
        __syncthreads();
    }
    if (t == 0) rowstart[NNODES] = s_carry;   // == EPRIME
}

__global__ __launch_bounds__(256)
void k_csr_fill(const int* __restrict__ ei, const int* __restrict__ rowstart,
                int* __restrict__ cursor, int* __restrict__ adj_src) {
    int e = blockIdx.x * 256 + threadIdx.x;
    if (e >= EPRIME) return;
    int s, d; edge_endpoints(ei, e, s, d);
    int pos = atomicAdd(&cursor[d], 1);
    adj_src[rowstart[d] + pos] = s;
}

// ---------------------------------------------------------------- xh = h @ W  (WMMA f16->f32)
__global__ __launch_bounds__(256)
void k_gemm_xh(const float* __restrict__ h, const float* __restrict__ W,
               float* __restrict__ xh) {
    __shared__ _Float16 sW[DIM * LDH];   // sW[n*LDH + k]  (transposed W)
    __shared__ _Float16 sA[16 * LDH];    // sA[r*LDH + k]
    const int t    = threadIdx.x;
    const int row0 = blockIdx.x * 16;

    for (int i = t; i < DIM * DIM; i += 256) {
        int k = i >> 7, n = i & 127;
        sW[n * LDH + k] = (_Float16)W[i];
    }
    for (int i = t; i < 16 * DIM; i += 256) {
        int r = i >> 7, c = i & 127;
        sA[r * LDH + c] = (_Float16)h[(row0 + r) * DIM + c];
    }
    __syncthreads();

    const int lane = t & 31;
    const int n0   = (t >> 5) * 16;
    const int mA   = lane & 15;
    const int kg   = lane >> 4;
    const int nB   = n0 + (lane & 15);

    v8f acc = {};
    #pragma unroll
    for (int kk = 0; kk < 4; ++kk) {
        const int kbase = kk * 32;
        v8h a0 = *(const v8h*)&sA[mA * LDH + kbase + kg * 8];
        v8h a1 = *(const v8h*)&sA[mA * LDH + kbase + 16 + kg * 8];
        v16h a = __builtin_shufflevector(a0, a1, 0,1,2,3,4,5,6,7,8,9,10,11,12,13,14,15);
        v8h b0 = *(const v8h*)&sW[nB * LDH + kbase + kg * 8];
        v8h b1 = *(const v8h*)&sW[nB * LDH + kbase + 16 + kg * 8];
        v16h b = __builtin_shufflevector(b0, b1, 0,1,2,3,4,5,6,7,8,9,10,11,12,13,14,15);
        acc = __builtin_amdgcn_wmma_f32_16x16x32_f16(false, a, false, b,
                                                     (short)0, acc, false, false);
    }
    const int col   = n0 + (lane & 15);
    const int rbase = row0 + (lane >> 4) * 8;
    #pragma unroll
    for (int r = 0; r < 8; ++r)
        xh[(rbase + r) * DIM + col] = acc[r];
}

// ---------------------------------------------------------------- per-(node,head) attention logits
__global__ __launch_bounds__(256)
void k_attn_node(const float* __restrict__ xh, const float* __restrict__ asrc,
                 const float* __restrict__ adst, float* __restrict__ als,
                 float* __restrict__ ald) {
    int i = blockIdx.x * 256 + threadIdx.x;      // i = n*NH + h
    if (i >= NNODES * NH) return;
    int n = i >> 2, hh = i & 3;
    const float* v  = xh + n * DIM + hh * DPH;
    const float* as = asrc + hh * DPH;
    const float* ad = adst + hh * DPH;
    float s = 0.f, d = 0.f;
    #pragma unroll
    for (int k = 0; k < DPH; ++k) { s += v[k] * as[k]; d += v[k] * ad[k]; }
    als[i] = s; ald[i] = d;
}

// ---------------------------------------------------------------- per-node segment softmax (wave per node)
// 8 lanes per head stride over the node's incoming-edge bucket; shfl reductions.
__global__ __launch_bounds__(256)
void k_softmax(const int* __restrict__ rowstart, const int* __restrict__ adj_src,
               const float* __restrict__ als, const float* __restrict__ ald,
               float* __restrict__ alpha) {
    int node = blockIdx.x * 8 + (threadIdx.x >> 5);
    if (node >= NNODES) return;
    int lane = threadIdx.x & 31;
    int hh  = lane >> 3;     // head
    int sub = lane & 7;      // position within 8-lane head group
    int beg = rowstart[node], end = rowstart[node + 1];
    float ald_n = ald[node * 4 + hh];

    float mx = -INFINITY;
    for (int p = beg + sub; p < end; p += 8) {
        int s = adj_src[p];
        float e = als[s * 4 + hh] + ald_n;
        e = (e > 0.f) ? e : NEG_SLOPE * e;
        mx = fmaxf(mx, e);
    }
    #pragma unroll
    for (int off = 4; off >= 1; off >>= 1) mx = fmaxf(mx, __shfl_xor(mx, off, 32));

    float sum = 0.f;
    for (int p = beg + sub; p < end; p += 8) {
        int s = adj_src[p];
        float e = als[s * 4 + hh] + ald_n;
        e = (e > 0.f) ? e : NEG_SLOPE * e;
        sum += __expf(e - mx);
    }
    #pragma unroll
    for (int off = 4; off >= 1; off >>= 1) sum += __shfl_xor(sum, off, 32);
    float inv = 1.0f / (sum + 1e-16f);

    for (int p = beg + sub; p < end; p += 8) {
        int s = adj_src[p];
        float e = als[s * 4 + hh] + ald_n;
        e = (e > 0.f) ? e : NEG_SLOPE * e;
        alpha[p * 4 + hh] = __expf(e - mx) * inv;
    }
}

// ---------------------------------------------------------------- aggregation: pure gather, no atomics (wave per node)
__global__ __launch_bounds__(256)
void k_aggr(const int* __restrict__ rowstart, const int* __restrict__ adj_src,
            const float* __restrict__ xh, const float* __restrict__ alpha,
            float* __restrict__ hnew) {
    int node = blockIdx.x * 8 + (threadIdx.x >> 5);
    if (node >= NNODES) return;
    int lane = threadIdx.x & 31;
    int hh = lane >> 3;                  // dims lane*4..lane*4+3 belong to head hh
    int beg = rowstart[node], end = rowstart[node + 1];
    float4 acc = make_float4(0.f, 0.f, 0.f, 0.f);
    for (int p = beg; p < end; ++p) {
        int s = adj_src[p];
        float a = alpha[p * 4 + hh];
        const float4 xv = *(const float4*)(xh + s * DIM + lane * 4);  // coalesced 512B row
        acc.x += a * xv.x; acc.y += a * xv.y;
        acc.z += a * xv.z; acc.w += a * xv.w;
    }
    *(float4*)(hnew + node * DIM + lane * 4) = acc;
}

// ---------------------------------------------------------------- bias + LayerNorm + (ReLU) + residual (wave per node)
__global__ __launch_bounds__(256)
void k_finish(const float* __restrict__ hnew, const float* __restrict__ cb,
              const float* __restrict__ g, const float* __restrict__ lb,
              const float* __restrict__ hin, float* __restrict__ hout, int do_relu) {
    int node = blockIdx.x * 8 + (threadIdx.x >> 5);
    if (node >= NNODES) return;
    int lane = threadIdx.x & 31;
    const float4 v4  = *(const float4*)(hnew + node * DIM + lane * 4);
    const float4 cb4 = *(const float4*)(cb + lane * 4);
    float v[4] = { v4.x + cb4.x, v4.y + cb4.y, v4.z + cb4.z, v4.w + cb4.w };

    float sum = v[0] + v[1] + v[2] + v[3];
    #pragma unroll
    for (int off = 16; off >= 1; off >>= 1) sum += __shfl_xor(sum, off, 32);
    float mu = sum * (1.0f / DIM);

    float sq = 0.f;
    #pragma unroll
    for (int k = 0; k < 4; ++k) { float dv = v[k] - mu; sq += dv * dv; }
    #pragma unroll
    for (int off = 16; off >= 1; off >>= 1) sq += __shfl_xor(sq, off, 32);
    float inv = rsqrtf(sq * (1.0f / DIM) + LN_EPS);

    const float4 g4  = *(const float4*)(g  + lane * 4);
    const float4 lb4 = *(const float4*)(lb + lane * 4);
    const float4 h4  = *(const float4*)(hin + node * DIM + lane * 4);
    float gg[4] = { g4.x, g4.y, g4.z, g4.w };
    float bb[4] = { lb4.x, lb4.y, lb4.z, lb4.w };
    float hh[4] = { h4.x, h4.y, h4.z, h4.w };
    float out[4];
    #pragma unroll
    for (int k = 0; k < 4; ++k) {
        float y = (v[k] - mu) * inv * gg[k] + bb[k];
        if (do_relu) y = fmaxf(y, 0.f);
        out[k] = hh[k] + y;
    }
    *(float4*)(hout + node * DIM + lane * 4) = make_float4(out[0], out[1], out[2], out[3]);
}

// ---------------------------------------------------------------- host side
extern "C" void kernel_launch(void* const* d_in, const int* in_sizes, int n_in,
                              void* d_out, int out_size, void* d_ws, size_t ws_size,
                              hipStream_t stream) {
    const float* x    = (const float*)d_in[0];
    const float* w_in = (const float*)d_in[1];
    const float* b_in = (const float*)d_in[2];
    const float* W[3]  = { (const float*)d_in[3],  (const float*)d_in[9],  (const float*)d_in[15] };
    const float* AS[3] = { (const float*)d_in[4],  (const float*)d_in[10], (const float*)d_in[16] };
    const float* AD[3] = { (const float*)d_in[5],  (const float*)d_in[11], (const float*)d_in[17] };
    const float* CB[3] = { (const float*)d_in[6],  (const float*)d_in[12], (const float*)d_in[18] };
    const float* G[3]  = { (const float*)d_in[7],  (const float*)d_in[13], (const float*)d_in[19] };
    const float* LB[3] = { (const float*)d_in[8],  (const float*)d_in[14], (const float*)d_in[20] };
    const int*   ei    = (const int*)d_in[21];
    float* out = (float*)d_out;

    // workspace layout
    float* ws    = (float*)d_ws;
    float* h     = ws;                        // N*D floats
    float* xh    = h     + NNODES * DIM;      // N*D
    float* hnew  = xh    + NNODES * DIM;      // N*D
    float* als   = hnew  + NNODES * DIM;      // N*H
    float* ald   = als   + NNODES * NH;       // N*H
    float* alpha = ald   + NNODES * NH;       // EPRIME*H
    int*   deg      = (int*)(alpha + (size_t)EPRIME * NH); // N
    int*   cursor   = deg + NNODES;                        // N
    int*   rowstart = cursor + NNODES;                     // N+1
    int*   adj_src  = rowstart + NNODES + 1;               // EPRIME

    const int ND_BLK   = (NNODES * DIM + 255) / 256;
    const int NH_BLK   = (NNODES * NH + 255) / 256;
    const int N_BLK    = (NNODES + 255) / 256;
    const int EDGE_BLK = (EPRIME + 255) / 256;
    const int NODE_BLK = (NNODES + 7) / 8;     // wave-per-node kernels

    // h0 projection + CSR build (edge structure shared by all 3 layers)
    k_input<<<ND_BLK, 256, 0, stream>>>(x, w_in, b_in, h);
    k_csr_init<<<N_BLK, 256, 0, stream>>>(deg, cursor);
    k_csr_count<<<EDGE_BLK, 256, 0, stream>>>(ei, deg);
    k_csr_scan<<<1, 1024, 0, stream>>>(deg, rowstart);
    k_csr_fill<<<EDGE_BLK, 256, 0, stream>>>(ei, rowstart, cursor, adj_src);

    for (int i = 0; i < 3; ++i) {
        k_gemm_xh<<<NNODES / 16, 256, 0, stream>>>(h, W[i], xh);
        k_attn_node<<<NH_BLK, 256, 0, stream>>>(xh, AS[i], AD[i], als, ald);
        k_softmax<<<NODE_BLK, 256, 0, stream>>>(rowstart, adj_src, als, ald, alpha);
        k_aggr<<<NODE_BLK, 256, 0, stream>>>(rowstart, adj_src, xh, alpha, hnew);
        float* dst = (i < 2) ? h : out;
        k_finish<<<NODE_BLK, 256, 0, stream>>>(hnew, CB[i], G[i], LB[i], h, dst, (i < 2) ? 1 : 0);
    }
}
